// Node_only_75900662055232
// MI455X (gfx1250) — compile-verified, matrix operand order checked by reading, not run on hardware
//
#include <hip/hip_runtime.h>

// ---- vector types matching the CDNA5 WMMA operand shapes ----
typedef __attribute__((ext_vector_type(16))) _Float16 v16h;
typedef __attribute__((ext_vector_type(8)))  _Float16 v8h;
typedef __attribute__((ext_vector_type(8)))  float    v8f;

// ---------------------------------------------------------------------------
// utility kernels
// ---------------------------------------------------------------------------
__global__ void k_zero_f32(float* __restrict__ p, long long n) {
    long long i = (long long)blockIdx.x * blockDim.x + threadIdx.x;
    long long s = (long long)gridDim.x * blockDim.x;
    for (; i < n; i += s) p[i] = 0.0f;
}

__global__ void k_degree(const int* __restrict__ col, float* __restrict__ deg, int E) {
    int e = blockIdx.x * blockDim.x + threadIdx.x;
    if (e < E) atomicAdd(&deg[col[e]], 1.0f);
}

__global__ void k_dinv(const float* __restrict__ deg, float* __restrict__ dinv, int N) {
    int i = blockIdx.x * blockDim.x + threadIdx.x;
    if (i < N) dinv[i] = rsqrtf(deg[i] + 1.0f);   // +1 = self loop; always > 0
}

// x [N][Kin] f32  ->  xh [N][Kp] f16 (zero padded)
__global__ void k_cvt_x(const float* __restrict__ x, _Float16* __restrict__ xh,
                        int N, int Kin, int Kp) {
    long long total = (long long)N * Kp;
    long long i = (long long)blockIdx.x * blockDim.x + threadIdx.x;
    long long s = (long long)gridDim.x * blockDim.x;
    for (; i < total; i += s) {
        int n = (int)(i / Kp), k = (int)(i % Kp);
        xh[i] = (k < Kin) ? (_Float16)x[(long long)n * Kin + k] : (_Float16)0.0f;
    }
}

// W [Kin][Nout] f32  ->  Wt [Np][Kp] f16 transposed, zero padded
__global__ void k_wt(const float* __restrict__ W, _Float16* __restrict__ Wt,
                     int Kin, int Nout, int Kp, int Np) {
    long long total = (long long)Np * Kp;
    long long i = (long long)blockIdx.x * blockDim.x + threadIdx.x;
    long long s = (long long)gridDim.x * blockDim.x;
    for (; i < total; i += s) {
        int nc = (int)(i / Kp), k = (int)(i % Kp);
        Wt[i] = (nc < Nout && k < Kin) ? (_Float16)W[(long long)k * Nout + nc]
                                       : (_Float16)0.0f;
    }
}

// ---------------------------------------------------------------------------
// WMMA GEMM: C[M][Np] = A[M][Kp](f16) * Bt[Np][Kp](f16, pre-transposed W)
//
// Block = 8 waves, block tile = 16 rows x 256 cols.  Np is always a multiple
// of 32, so each wave owns a fully-valid 16x32 tile (2 WMMA accumulators) and
// the K loop is branch-free: one LDS A fragment (ds_load_b128) feeds two
// v_wmma issues per 32-wide K step.  The 16xKp A tile is contiguous in memory
// -> one cooperative flat copy into LDS per block.
//
// Per-lane fragment layout (16x16x32 f16, wave32):
//   A: lane<16 -> M=lane,   K in {kb+0..7, kb+16..23}
//      lane>=16-> M=lane-16,K in {kb+8..15, kb+24..31}
//   B: same K mapping, N = lane%16 (column of W == row of Wt)
//   C: VGPR v -> M = v + 8*(lane/16), N = lane%16
// ---------------------------------------------------------------------------
__global__ void k_gemm_wmma(const _Float16* __restrict__ A,
                            const _Float16* __restrict__ Bt,
                            float* __restrict__ C,
                            int M, int Kp, int Np) {
    __shared__ __align__(16) _Float16 smA[16 * 512];   // 16 KB max (Kp <= 512)

    const int lane = threadIdx.x & 31;
    const int wv   = threadIdx.x >> 5;       // 0..7
    const int tm   = blockIdx.x;             // 16-row tile

    // cooperative flat copy: rows tm*16 .. tm*16+15 are contiguous (stride Kp)
    {
        const v8h* src = (const v8h*)(A + (size_t)tm * 16 * Kp);
        v8h* dst = (v8h*)smA;
        const int nchunk = 2 * Kp;           // 16*Kp / 8 halves per chunk
        for (int i = threadIdx.x; i < nchunk; i += blockDim.x) dst[i] = src[i];
    }
    __syncthreads();

    const int n0 = (blockIdx.y * 8 + wv) * 32;   // this wave's 32-col group
    if (n0 >= Np) return;                        // whole-wave exit, after barrier

    const int r    = lane & 15;
    const int half = lane >> 4;

    const _Float16* arL = smA + (size_t)r * Kp;            // LDS row
    const _Float16* br0 = Bt + (size_t)(n0 + r) * Kp;
    const _Float16* br1 = Bt + (size_t)(n0 + 16 + r) * Kp;

    v8f acc0 = {}, acc1 = {};
    for (int kb = 0; kb < Kp; kb += 32) {
        __builtin_prefetch(br0 + kb + 32, 0, 3);   // global_prefetch_b8
        __builtin_prefetch(br1 + kb + 32, 0, 3);
        const int k0 = kb + 8 * half;
        const int k1 = kb + 16 + 8 * half;

        v8h alo = *(const v8h*)(arL + k0);   // ds_load_b128
        v8h ahi = *(const v8h*)(arL + k1);
        v16h a = __builtin_shufflevector(alo, ahi, 0,1,2,3,4,5,6,7,8,9,10,11,12,13,14,15);

        v8h b0lo = *(const v8h*)(br0 + k0);
        v8h b0hi = *(const v8h*)(br0 + k1);
        v16h b0 = __builtin_shufflevector(b0lo, b0hi, 0,1,2,3,4,5,6,7,8,9,10,11,12,13,14,15);
        acc0 = __builtin_amdgcn_wmma_f32_16x16x32_f16(
            false, a, false, b0, (short)0, acc0, false, false);

        v8h b1lo = *(const v8h*)(br1 + k0);
        v8h b1hi = *(const v8h*)(br1 + k1);
        v16h b1 = __builtin_shufflevector(b1lo, b1hi, 0,1,2,3,4,5,6,7,8,9,10,11,12,13,14,15);
        acc1 = __builtin_amdgcn_wmma_f32_16x16x32_f16(
            false, a, false, b1, (short)0, acc1, false, false);
    }

    const int mb = tm * 16 + 8 * half;
    float* c0 = C + (size_t)mb * Np + (n0 + r);
#pragma unroll
    for (int v = 0; v < 8; ++v) c0[(size_t)v * Np] = acc0[v];
    float* c1 = c0 + 16;
#pragma unroll
    for (int v = 0; v < 8; ++v) c1[(size_t)v * Np] = acc1[v];
}

// ---------------------------------------------------------------------------
// edge scatter: Agg[col[e]] += H[row[e]] * dinv[row]*dinv[col]  (float4 loads,
// f32 atomics resolving in L2; virtual edges e >= E are self loops)
// ---------------------------------------------------------------------------
__global__ void k_scatter(const int* __restrict__ row, const int* __restrict__ col,
                          const float* __restrict__ dinv,
                          const float* __restrict__ H, float* __restrict__ Agg,
                          int E, int N, int Np) {
    const int chunks = Np >> 2;
    const long long total = (long long)(E + N) * chunks;
    long long i = (long long)blockIdx.x * blockDim.x + threadIdx.x;
    long long s = (long long)gridDim.x * blockDim.x;
    for (; i < total; i += s) {
        long long e = i / chunks;
        int f = (int)(i - e * (long long)chunks) << 2;
        int r, c;
        if (e < E) { r = row[e]; c = col[e]; }
        else       { r = c = (int)(e - E); }
        float w = dinv[r] * dinv[c];
        float4 h = *(const float4*)(H + (size_t)r * Np + f);
        float* d = Agg + (size_t)c * Np + f;
        atomicAdd(d + 0, h.x * w);
        atomicAdd(d + 1, h.y * w);
        atomicAdd(d + 2, h.z * w);
        atomicAdd(d + 3, h.w * w);
    }
}

// mid layers: feat_f16 = relu(Agg + b)   (Nout == Np, also next layer's Kp)
__global__ void k_bias_relu_f16(const float* __restrict__ Agg, const float* __restrict__ b,
                                _Float16* __restrict__ feat, int N, int Np) {
    long long total = (long long)N * Np;
    long long i = (long long)blockIdx.x * blockDim.x + threadIdx.x;
    long long s = (long long)gridDim.x * blockDim.x;
    for (; i < total; i += s) {
        int f = (int)(i % Np);
        feat[i] = (_Float16)fmaxf(Agg[i] + b[f], 0.0f);
    }
}

// final layer: out[f] += relu(Agg + b) / N   (d_out zeroed beforehand)
__global__ void k_final_mean(const float* __restrict__ Agg, const float* __restrict__ b,
                             float* __restrict__ out, int N, int Np, int Nout) {
    long long total = (long long)N * Nout;
    long long i = (long long)blockIdx.x * blockDim.x + threadIdx.x;
    long long s = (long long)gridDim.x * blockDim.x;
    const float inv = 1.0f / (float)N;
    for (; i < total; i += s) {
        int n = (int)(i / Nout), f = (int)(i % Nout);
        float v = fmaxf(Agg[(size_t)n * Np + f] + b[f], 0.0f);
        atomicAdd(&out[f], v * inv);
    }
}

// ---------------------------------------------------------------------------
extern "C" void kernel_launch(void* const* d_in, const int* in_sizes, int n_in,
                              void* d_out, int out_size, void* d_ws, size_t ws_size,
                              hipStream_t stream) {
    const float* x  = (const float*)d_in[0];
    const int*   ei = (const int*)d_in[1];
    const int N = in_sizes[0] / 100;   // 50000, divisible by 16
    const int E = in_sizes[1] / 2;     // 800000
    const int* row = ei;               // sources
    const int* col = ei + E;           // targets

    // layer geometry; all padded widths are multiples of 32 so the WMMA
    // K-loop needs no subtile guard (pad cols of Wt are zero -> outputs zero)
    const int Kin[4]  = {100, 512, 256, 128};
    const int Kp[4]   = {128, 512, 256, 128};
    const int Nout[4] = {512, 256, 128, 200};
    const int Np[4]   = {512, 256, 128, 224};

    // workspace carve-up
    char* ws = (char*)d_ws;
    size_t off = 0;
    float* deg  = (float*)(ws + off); off += (((size_t)N * 4) + 255) & ~(size_t)255;
    float* dinv = (float*)(ws + off); off += (((size_t)N * 4) + 255) & ~(size_t)255;
    _Float16* Wt   = (_Float16*)(ws + off); off += 512 * 1024;          // >= 256*512*2
    _Float16* feat = (_Float16*)(ws + off); off += (size_t)N * 512 * 2; // f16 activations
    float*    H    = (float*)(ws + off);    off += (size_t)N * 512 * 4; // GEMM output
    float*    Agg  = (float*)(ws + off);    off += (size_t)N * 512 * 4; // scatter target

    // ---- normalization: deg -> dinv ----
    k_zero_f32<<<256, 256, 0, stream>>>(deg, N);
    k_degree<<<(E + 255) / 256, 256, 0, stream>>>(col, deg, E);
    k_dinv<<<(N + 255) / 256, 256, 0, stream>>>(deg, dinv, N);

    // ---- input features -> padded f16 ----
    k_cvt_x<<<2048, 256, 0, stream>>>(x, feat, N, Kin[0], Kp[0]);

    // ---- 4 GCN layers ----
    for (int l = 0; l < 4; ++l) {
        const float* W = (const float*)d_in[2 + 2 * l];
        const float* b = (const float*)d_in[3 + 2 * l];

        long long wtn = (long long)Np[l] * Kp[l];
        k_wt<<<(unsigned)((wtn + 255) / 256), 256, 0, stream>>>(W, Wt, Kin[l], Nout[l], Kp[l], Np[l]);

        dim3 g(N / 16, (Np[l] + 255) / 256);   // block tile = 16 x 256
        k_gemm_wmma<<<g, 256, 0, stream>>>(feat, Wt, H, N, Kp[l], Np[l]);

        k_zero_f32<<<2048, 256, 0, stream>>>(Agg, (long long)N * Np[l]);
        k_scatter<<<8192, 256, 0, stream>>>(row, col, dinv, H, Agg, E, N, Np[l]);

        if (l < 3) {
            k_bias_relu_f16<<<2048, 256, 0, stream>>>(Agg, b, feat, N, Np[l]);
        } else {
            k_zero_f32<<<1, 256, 0, stream>>>((float*)d_out, out_size);
            k_final_mean<<<2048, 256, 0, stream>>>(Agg, b, (float*)d_out, N, Np[l], Nout[l]);
        }
    }
}
